// FreqFusion_42442866819155
// MI455X (gfx1250) — matrix operand order repaired and checked by reading, not compiled, not run on hardware
//
#include <hip/hip_runtime.h>

// FreqFusion for MI455X (gfx1250, wave32).
// Conv stages are WMMA f16->f32 GEMMs (v_wmma_f32_16x16x32_f16) with weights
// pre-packed to f16 (zero-padded [NT*16, Kpad]) and staged in LDS via the
// CDNA5 async-to-LDS path (ASYNCcnt) when available; B fragments are
// 2x ds_load_b128 each. Everything else streams fp32 at HBM/L2 bandwidth.

typedef __attribute__((ext_vector_type(16))) _Float16     v16h;
typedef __attribute__((ext_vector_type(8)))  float        v8f;
typedef __attribute__((ext_vector_type(4)))  float        f32x4;
typedef __attribute__((ext_vector_type(4)))  unsigned int u32x4;

#if defined(__HIP_DEVICE_COMPILE__) && \
    __has_builtin(__builtin_amdgcn_global_load_async_to_lds_b128) && \
    __has_builtin(__builtin_amdgcn_s_wait_asynccnt)
#define HAS_ASYNC_LDS 1
// Builtin signature (from compiler diagnostic): param0 = int4 vector ptr in
// AS1 (global), param1 = int4 vector ptr in AS3 (LDS), then imm offset, cpol.
typedef int v4i_async __attribute__((vector_size(16)));
typedef __attribute__((address_space(1))) v4i_async* gv4i_t;
typedef __attribute__((address_space(3))) v4i_async* lv4i_t;
#endif

#define B_N  2
#define HWH  16384   // 128*128
#define HWL  4096    // 64*64

__device__ __forceinline__ float hamw(int i, int n) {
  return 0.54f - 0.46f * __cosf(6.28318530717958647692f * (float)i / (float)(n - 1));
}

// ---------------------------------------------------------------------------
// Pack f32 OIHW weights -> f16 [Coutpad, Kpad] with zero padding.
// ---------------------------------------------------------------------------
__global__ void wpack_k(const float* __restrict__ w, _Float16* __restrict__ dst,
                        int Cout, int K, int Kpad, int total) {
  int idx = blockIdx.x * blockDim.x + threadIdx.x;
  if (idx >= total) return;
  int n = idx / Kpad;
  int k = idx - n * Kpad;
  float v = (n < Cout && k < K) ? w[n * K + k] : 0.0f;
  dst[idx] = (_Float16)v;
}

// ---------------------------------------------------------------------------
// Generic conv (ksize 1 or 3, pad ks/2) as WMMA GEMM. H == W == 1<<logW.
// out[b,n,p] = bias[n] + sum_k im2col(in)[p,k] * w[n,k]
// One wave computes 16 pixels x 16*NT out-channels. Grids sized exactly so
// every wave is full (EXEC all ones at each WMMA).
// ---------------------------------------------------------------------------
template <int NT, int KS>
__global__ void conv_wmma_k(const float* __restrict__ in, const _Float16* __restrict__ wpad,
                            const float* __restrict__ bias, float* __restrict__ out,
                            int Cin, int Cout, int logW) {
  const int W      = 1 << logW;
  const int H      = W;
  const int HW     = 1 << (2 * logW);
  const int K      = Cin * KS * KS;
  const int Ksteps = (K + 31) >> 5;
  const int Kpad   = Ksteps << 5;

  // Stage padded f16 weights into LDS (async-to-LDS when the toolchain has it).
  __shared__ alignas(16) _Float16 wlds[18432];   // 36 KB max (NT*16*Kpad)
  {
    const int copyN = (NT * 16 * Kpad) >> 3;     // number of 16-byte chunks
    const u32x4* lsrc = (const u32x4*)wpad;
    u32x4* ldst = (u32x4*)wlds;
#if defined(HAS_ASYNC_LDS)
    for (int i = threadIdx.x; i < copyN; i += blockDim.x)
      __builtin_amdgcn_global_load_async_to_lds_b128(
          (gv4i_t)(lsrc + i), (lv4i_t)(ldst + i), 0, 0);
    __builtin_amdgcn_s_wait_asynccnt(0);
#else
    for (int i = threadIdx.x; i < copyN; i += blockDim.x) ldst[i] = lsrc[i];
#endif
  }
  __syncthreads();

  const int lane   = threadIdx.x & 31;
  const int waveId = threadIdx.x >> 5;
  const int mtile  = blockIdx.x * (blockDim.x >> 5) + waveId;
  const int b      = mtile >> (2 * logW - 4);
  const int p0     = (mtile << 4) & (HW - 1);
  const int lrow   = lane & 15;
  const int khalfA = (lane >> 4) << 3;   // D frag M (pixel) offset: 0 or 8
  const int khalfB = (lane >> 4) << 4;   // B frag K offset: 0 or 16
  const int p      = p0 + lrow;
  const int y      = p >> logW;
  const int x      = p & (W - 1);
  const int pad    = KS >> 1;

  v8f acc[NT];
  for (int t = 0; t < NT; ++t)
    for (int e = 0; e < 8; ++e) acc[t][e] = 0.0f;

  const float* inb = in + (size_t)b * Cin * HW;
  union Frag { v16h v; _Float16 h[16]; u32x4 q[2]; };

  for (int kstep = 0; kstep < Ksteps; ++kstep) {
    const int k0 = kstep << 5;
    Frag A;
#pragma unroll
    for (int j = 0; j < 16; ++j) {
      int kidx = k0 + khalfB + j;
      float v = 0.0f;
      if (kidx < K) {
        int c  = kidx / (KS * KS);           // constant divisor (template)
        int t  = kidx - c * (KS * KS);
        int ty = t / KS;
        int dy = ty - pad;
        int dx = t - ty * KS - pad;
        int yy = y + dy, xx = x + dx;
        if (yy >= 0 && yy < H && xx >= 0 && xx < W)
          v = inb[(size_t)c * HW + (yy << logW) + xx];
      }
      A.h[j] = (_Float16)v;
    }
#pragma unroll
    for (int nt = 0; nt < NT; ++nt) {
      int n = (nt << 4) + lrow;              // padded rows are zero-filled
      Frag Bf;
      const u32x4* bp = (const u32x4*)(wlds + n * Kpad + k0 + khalfB);
      Bf.q[0] = bp[0];
      Bf.q[1] = bp[1];
      acc[nt] = __builtin_amdgcn_wmma_f32_16x16x32_f16(
          false, A.v, false, Bf.v, (short)0, acc[nt], false, false);
    }
  }

#pragma unroll
  for (int nt = 0; nt < NT; ++nt) {
    int n = (nt << 4) + lrow;
    if (n >= Cout) continue;
    float bv = bias[n];
    float* ob = out + ((size_t)b * Cout + n) * HW + p0 + khalfA;
    f32x4 lo, hi;
#pragma unroll
    for (int e = 0; e < 4; ++e) { lo[e] = acc[nt][e] + bv; hi[e] = acc[nt][e + 4] + bv; }
    *(f32x4*)(ob)     = lo;                  // 32B-aligned (p0 mult of 16)
    *(f32x4*)(ob + 4) = hi;
  }
}

// ---------------------------------------------------------------------------
// kernel_normalizer: softmax over KS*KS channels * hamming, renormalized.
// (softmax denominator cancels with the renormalization.)
// ---------------------------------------------------------------------------
template <int KS>
__global__ void knorm_k(const float* __restrict__ in, float* __restrict__ out,
                        int Bn, int H, int W) {
  const int K2 = KS * KS;
  const int HW = H * W;
  int idx = blockIdx.x * blockDim.x + threadIdx.x;
  if (idx >= Bn * HW) return;
  int b = idx / HW;
  int p = idx - b * HW;
  const float* ip = in + (size_t)b * K2 * HW + p;
  float v[K2];
  float m = -1e30f;
#pragma unroll
  for (int i = 0; i < K2; ++i) { v[i] = ip[(size_t)i * HW]; m = fmaxf(m, v[i]); }
  float s2 = 0.0f;
#pragma unroll
  for (int i = 0; i < K2; ++i) {
    float hv = hamw(i / KS, KS) * hamw(i - (i / KS) * KS, KS);
    v[i] = __expf(v[i] - m) * hv;
    s2 += v[i];
  }
  float inv = 1.0f / s2;
  float* op = out + (size_t)b * K2 * HW + p;
#pragma unroll
  for (int i = 0; i < K2; ++i) op[(size_t)i * HW] = v[i] * inv;
}

// ---------------------------------------------------------------------------
// out = 2*res - carafe(feat, mask, k=3, scale=1)
// ---------------------------------------------------------------------------
__global__ void carafe1_res_k(const float* __restrict__ feat, const float* __restrict__ mask,
                              const float* __restrict__ res, float* __restrict__ out,
                              int Bn, int C, int H, int W) {
  const int HW = H * W;
  int idx = blockIdx.x * blockDim.x + threadIdx.x;
  if (idx >= Bn * C * HW) return;
  int b = idx / (C * HW);
  int r = idx - b * C * HW;
  int c = r / HW;
  int p = r - c * HW;
  int h = p / W, w = p - (p / W) * W;
  const float* fb = feat + ((size_t)b * C + c) * HW;
  const float* mb = mask + (size_t)b * 9 * HW + p;
  float s = 0.0f;
#pragma unroll
  for (int i = 0; i < 9; ++i) {
    int dy = i / 3 - 1, dx = i - (i / 3) * 3 - 1;
    int yy = h + dy, xx = w + dx;
    float fv = (yy >= 0 && yy < H && xx >= 0 && xx < W) ? fb[yy * W + xx] : 0.0f;
    s += fv * mb[(size_t)i * HW];
  }
  out[idx] = 2.0f * res[idx] - s;
}

// ---------------------------------------------------------------------------
// carafe scale=2, k=5. mask on 2H x 2W grid; optional elementwise addend.
// ---------------------------------------------------------------------------
__global__ void carafe2_k(const float* __restrict__ feat, const float* __restrict__ mask,
                          const float* __restrict__ addend, float* __restrict__ out,
                          int Bn, int C, int Hl, int Wl) {
  const int Ho = 2 * Hl, Wo = 2 * Wl;
  const int HWo = Ho * Wo;
  int idx = blockIdx.x * blockDim.x + threadIdx.x;
  if (idx >= Bn * C * HWo) return;
  int b = idx / (C * HWo);
  int r = idx - b * C * HWo;
  int c = r / HWo;
  int p = r - c * HWo;
  int Y = p / Wo, X = p - (p / Wo) * Wo;
  int h = Y >> 1, w = X >> 1;
  const float* fb = feat + ((size_t)b * C + c) * (size_t)Hl * Wl;
  const float* mb = mask + (size_t)b * 25 * HWo + p;
  float s = 0.0f;
#pragma unroll
  for (int i = 0; i < 25; ++i) {
    int dy = i / 5 - 2, dx = i - (i / 5) * 5 - 2;
    int yy = h + dy, xx = w + dx;
    float fv = (yy >= 0 && yy < Hl && xx >= 0 && xx < Wl) ? fb[yy * Wl + xx] : 0.0f;
    s += fv * mb[(size_t)i * HWo];
  }
  out[idx] = s + (addend ? addend[idx] : 0.0f);
}

// ---------------------------------------------------------------------------
// GroupNorm: one block per (group, batch). Biased variance, eps=1e-5.
// ---------------------------------------------------------------------------
__global__ void groupnorm_k(const float* __restrict__ in, const float* __restrict__ gamma,
                            const float* __restrict__ beta, float* __restrict__ out,
                            int C, int HW, int groups) {
  int g = blockIdx.x, b = blockIdx.y;
  int cpg = C / groups;
  size_t base = ((size_t)b * C + (size_t)g * cpg) * HW;
  size_t n = (size_t)cpg * HW;
  float s = 0.0f, ss = 0.0f;
  for (size_t i = threadIdx.x; i < n; i += blockDim.x) {
    float v = in[base + i]; s += v; ss += v * v;
  }
  __shared__ float rs[256], rss[256];
  rs[threadIdx.x] = s; rss[threadIdx.x] = ss;
  __syncthreads();
  for (int o = 128; o > 0; o >>= 1) {
    if ((int)threadIdx.x < o) { rs[threadIdx.x] += rs[threadIdx.x + o]; rss[threadIdx.x] += rss[threadIdx.x + o]; }
    __syncthreads();
  }
  float mean = rs[0] / (float)n;
  float var  = rss[0] / (float)n - mean * mean;
  float inv  = rsqrtf(var + 1e-5f);
  for (size_t i = threadIdx.x; i < n; i += blockDim.x) {
    int c = g * cpg + (int)(i / HW);
    out[base + i] = (in[base + i] - mean) * inv * gamma[c] + beta[c];
  }
}

// ---------------------------------------------------------------------------
// compute_similarity: 3x3 neighborhood, dilation 2, cosine sim, 8 outputs.
// ---------------------------------------------------------------------------
__global__ void similarity_k(const float* __restrict__ x, float* __restrict__ out,
                             int Bn, int C, int H, int W) {
  const int HW = H * W;
  int idx = blockIdx.x * blockDim.x + threadIdx.x;
  if (idx >= Bn * HW) return;
  int b = idx / HW;
  int p = idx - b * HW;
  int h = p / W, w = p - (p / W) * W;
  const float* xb = x + (size_t)b * C * HW;
  float cn = 0.0f;
  for (int c = 0; c < C; ++c) { float v = xb[(size_t)c * HW + p]; cn += v * v; }
  cn = fmaxf(sqrtf(cn), 1e-8f);
  int oc = 0;
#pragma unroll
  for (int i = 0; i < 9; ++i) {
    if (i == 4) continue;
    int dy = (i / 3) * 2 - 2, dx = (i - (i / 3) * 3) * 2 - 2;
    int yy = h + dy, xx = w + dx;
    float dot = 0.0f, nn = 0.0f;
    if (yy >= 0 && yy < H && xx >= 0 && xx < W) {
      int q = yy * W + xx;
      for (int c = 0; c < C; ++c) {
        float a = xb[(size_t)c * HW + p];
        float e = xb[(size_t)c * HW + q];
        dot += a * e; nn += e * e;
      }
    }
    out[((size_t)b * 8 + oc) * HW + p] = dot / (cn * fmaxf(sqrtf(nn), 1e-8f));
    ++oc;
  }
}

// ---------------------------------------------------------------------------
// offset = (off_lr + unshuffle(off_hr)) * sigmoid(ds_lr + unshuffle(ds_hr)) + init_pos
// ---------------------------------------------------------------------------
__global__ void offset_k(const float* __restrict__ offl, const float* __restrict__ offh,
                         const float* __restrict__ dsl, const float* __restrict__ dsh,
                         float* __restrict__ offset, int Bn, int H, int W) {
  const int HW = H * W;
  int idx = blockIdx.x * blockDim.x + threadIdx.x;
  if (idx >= Bn * 32 * HW) return;
  int b  = idx / (32 * HW);
  int r  = idx - b * 32 * HW;
  int ch = r / HW;
  int p  = r - ch * HW;
  int h = p / W, w = p - (p / W) * W;
  int c8 = ch >> 2, ii = (ch >> 1) & 1, jj = ch & 1;
  int H2 = 2 * H, W2 = 2 * W;
  size_t hidx = ((size_t)b * 8 + c8) * (size_t)H2 * W2 + (size_t)(2 * h + ii) * W2 + (2 * w + jj);
  float off = offl[idx] + offh[hidx];
  float ds  = dsl[idx] + dsh[hidx];
  float sig = 1.0f / (1.0f + __expf(-ds));
  int a = ch >> 4, g = ch & 15;
  float hv0 = -0.25f, hv1 = 0.25f;
  float ip = (a == 0) ? ((g & 1) ? hv1 : hv0) : (((g >> 1) & 1) ? hv1 : hv0);
  offset[idx] = off * sig + ip;
}

// ---------------------------------------------------------------------------
// DySample gather: bilinear grid_sample (border clamp) of lr_up per group.
// ---------------------------------------------------------------------------
__global__ void dysample_k(const float* __restrict__ offset, const float* __restrict__ lrup,
                           float* __restrict__ out, int Bn) {
  const int Ho = 128, Wo = 128, Hl = 64, Wl = 64;
  int idx = blockIdx.x * blockDim.x + threadIdx.x;
  if (idx >= Bn * 4 * Ho * Wo) return;
  int b = idx / (4 * Ho * Wo);
  int r = idx - b * 4 * Ho * Wo;
  int grp = r / (Ho * Wo);
  int p = r - grp * (Ho * Wo);
  int Y = p / Wo, X = p - (p / Wo) * Wo;
  int h = Y >> 1, w = X >> 1, ii = Y & 1, jj = X & 1;
  int g = grp * 4 + ii * 2 + jj;
  float offx = offset[((size_t)b * 32 + g) * Hl * Wl + h * Wl + w];
  float offy = offset[((size_t)b * 32 + 16 + g) * Hl * Wl + h * Wl + w];
  float gxn = 2.0f * ((float)w + 0.5f + offx) / (float)Wl - 1.0f;
  float gyn = 2.0f * ((float)h + 0.5f + offy) / (float)Hl - 1.0f;
  float gx = fminf(fmaxf((gxn + 1.0f) * (float)Wo * 0.5f - 0.5f, 0.0f), (float)(Wo - 1));
  float gy = fminf(fmaxf((gyn + 1.0f) * (float)Ho * 0.5f - 0.5f, 0.0f), (float)(Ho - 1));
  float x0f = floorf(gx), y0f = floorf(gy);
  float tx = gx - x0f, ty = gy - y0f;
  int x0 = (int)x0f, y0 = (int)y0f;
  int x1 = x0 + 1; if (x1 > Wo - 1) x1 = Wo - 1;
  int y1 = y0 + 1; if (y1 > Ho - 1) y1 = Ho - 1;
  float w00 = (1.0f - tx) * (1.0f - ty), w01 = tx * (1.0f - ty);
  float w10 = (1.0f - tx) * ty,          w11 = tx * ty;
  const float* src = lrup + ((size_t)b * 256 + (size_t)grp * 64) * Ho * Wo;
  float* dst = out + ((size_t)b * 256 + (size_t)grp * 64) * Ho * Wo + Y * Wo + X;
  for (int c = 0; c < 64; ++c) {
    const float* sc = src + (size_t)c * Ho * Wo;
    __builtin_prefetch(sc + Ho * Wo + y0 * Wo + x0, 0, 0);  // next channel plane
    float v = sc[y0 * Wo + x0] * w00 + sc[y0 * Wo + x1] * w01 +
              sc[y1 * Wo + x0] * w10 + sc[y1 * Wo + x1] * w11;
    dst[(size_t)c * Ho * Wo] = v;
  }
}

static inline int cdiv(int a, int b) { return (a + b - 1) / b; }

extern "C" void kernel_launch(void* const* d_in, const int* in_sizes, int n_in,
                              void* d_out, int out_size, void* d_ws, size_t ws_size,
                              hipStream_t stream) {
  (void)in_sizes; (void)n_in; (void)out_size; (void)ws_size;
  const float* hr_feat   = (const float*)d_in[0];
  const float* lr_feat   = (const float*)d_in[1];
  const float* w_hr_comp = (const float*)d_in[2];
  const float* b_hr_comp = (const float*)d_in[3];
  const float* w_lr_comp = (const float*)d_in[4];
  const float* b_lr_comp = (const float*)d_in[5];
  const float* w_enc     = (const float*)d_in[6];
  const float* b_enc     = (const float*)d_in[7];
  const float* w_enc2    = (const float*)d_in[8];
  const float* b_enc2    = (const float*)d_in[9];
  const float* gn_hr_w   = (const float*)d_in[10];
  const float* gn_hr_b   = (const float*)d_in[11];
  const float* gn_lr_w   = (const float*)d_in[12];
  const float* gn_lr_b   = (const float*)d_in[13];
  const float* w_off     = (const float*)d_in[14];
  const float* b_off     = (const float*)d_in[15];
  const float* w_hoff    = (const float*)d_in[16];
  const float* b_hoff    = (const float*)d_in[17];
  const float* w_ds      = (const float*)d_in[18];
  const float* b_ds      = (const float*)d_in[19];
  const float* w_hds     = (const float*)d_in[20];
  const float* b_hds     = (const float*)d_in[21];

  float* out_mask = (float*)d_out;                 // [2,25,128,128]
  float* out_hr   = out_mask + 819200;             // [2,256,128,128]
  float* out_lr   = out_hr + 8388608;              // [2,256,128,128]

  float* ws      = (float*)d_ws;
  float* ws_ch   = ws;                  // [2,64,128,128]
  float* ws_cl   = ws_ch   + 2097152;   // [2,64,64,64]
  float* ws_mhh  = ws_cl   + 524288;    // mask_hr_hr [2,9,128,128]
  float* ws_mhi  = ws_mhh  + 294912;    // mh_init
  float* ws_ch2  = ws_mhi  + 294912;    // updated ch
  float* ws_mlh  = ws_ch2  + 2097152;   // mask_lr_hr [2,25,128,128]
  float* ws_mli  = ws_mlh  + 819200;    // ml_init
  float* ws_encl = ws_mli  + 819200;    // conv(cl,w_enc) [2,25,64,64]
  float* ws_mlr  = ws_encl + 204800;    // mask_lr
  float* ws_enc2 = ws_mlr  + 819200;    // conv(cl,w_enc2) [2,9,64,64]
  float* ws_mhr  = ws_enc2 + 73728;     // mask_hr
  float* ws_mhn  = ws_mhr  + 294912;    // mask_hr_n
  float* ws_lrup = ws_mhn  + 294912;    // lr_up [2,256,128,128]
  float* ws_hrx  = ws_lrup + 8388608;   // group_norm(ch)
  float* ws_lrx  = ws_hrx  + 2097152;   // group_norm(cl)
  float* ws_hsim = ws_lrx  + 524288;    // [2,8,128,128]
  float* ws_lsim = ws_hsim + 262144;    // [2,8,64,64]
  float* ws_offl = ws_lsim + 65536;     // [2,32,64,64]
  float* ws_offh = ws_offl + 262144;    // [2,8,128,128]
  float* ws_dsl  = ws_offh + 262144;
  float* ws_dsh  = ws_dsl  + 262144;
  float* ws_offs = ws_dsh  + 262144;    // offset [2,32,64,64]

  // f16 packed-weight regions (each 16B aligned; sizes = Coutpad*Kpad)
  _Float16* p_whr   = (_Float16*)(ws_offs + 262144);
  _Float16* p_wlr   = p_whr   + 16384;  // 64x256
  _Float16* p_wenc  = p_wlr   + 16384;  // 32x576
  _Float16* p_wenc2 = p_wenc  + 18432;  // 16x576
  _Float16* p_woff  = p_wenc2 + 9216;   // 32x96
  _Float16* p_whoff = p_woff  + 3072;   // 16x96
  _Float16* p_wds   = p_whoff + 1536;   // 32x576
  _Float16* p_whds  = p_wds   + 18432;  // 16x576

  const int gHR = B_N * (HWH / 16) / 8; // 256 blocks (8 waves/block), exact
  const int gLR = B_N * (HWL / 16) / 8; // 64 blocks, exact

  // 0) Pack all conv weights to padded f16.
  wpack_k<<<cdiv(16384, 256), 256, 0, stream>>>(w_hr_comp, p_whr,  64, 256, 256, 16384);
  wpack_k<<<cdiv(16384, 256), 256, 0, stream>>>(w_lr_comp, p_wlr,  64, 256, 256, 16384);
  wpack_k<<<cdiv(18432, 256), 256, 0, stream>>>(w_enc,     p_wenc, 25, 576, 576, 18432);
  wpack_k<<<cdiv(9216,  256), 256, 0, stream>>>(w_enc2,    p_wenc2, 9, 576, 576, 9216);
  wpack_k<<<cdiv(3072,  256), 256, 0, stream>>>(w_off,     p_woff, 32,  72,  96, 3072);
  wpack_k<<<cdiv(1536,  256), 256, 0, stream>>>(w_hoff,    p_whoff, 8,  72,  96, 1536);
  wpack_k<<<cdiv(18432, 256), 256, 0, stream>>>(w_ds,      p_wds,  32, 576, 576, 18432);
  wpack_k<<<cdiv(9216,  256), 256, 0, stream>>>(w_hds,     p_whds,  8, 576, 576, 9216);

  // 1) 1x1 compressions (Cin=256 -> Cout=64), WMMA
  conv_wmma_k<4, 1><<<gHR, 256, 0, stream>>>(hr_feat, p_whr, b_hr_comp, ws_ch, 256, 64, 7);
  conv_wmma_k<4, 1><<<gLR, 256, 0, stream>>>(lr_feat, p_wlr, b_lr_comp, ws_cl, 256, 64, 6);

  // 2) mask_hr_hr = conv3x3(ch, w_enc2) ; mh_init ; ch = 2*ch - carafe(ch,mh,3,1)
  conv_wmma_k<1, 3><<<gHR, 256, 0, stream>>>(ws_ch, p_wenc2, b_enc2, ws_mhh, 64, 9, 7);
  knorm_k<3><<<cdiv(B_N * HWH, 256), 256, 0, stream>>>(ws_mhh, ws_mhi, B_N, 128, 128);
  carafe1_res_k<<<cdiv(B_N * 64 * HWH, 256), 256, 0, stream>>>(ws_ch, ws_mhi, ws_ch, ws_ch2, B_N, 64, 128, 128);

  // 3) mask_lr_hr = conv3x3(ch2, w_enc) ; ml_init
  conv_wmma_k<2, 3><<<gHR, 256, 0, stream>>>(ws_ch2, p_wenc, b_enc, ws_mlh, 64, 25, 7);
  knorm_k<5><<<cdiv(B_N * HWH, 256), 256, 0, stream>>>(ws_mlh, ws_mli, B_N, 128, 128);

  // 4) mask_lr = mask_lr_hr + carafe(conv3x3(cl,w_enc), ml_init, 5, 2)
  conv_wmma_k<2, 3><<<gLR, 256, 0, stream>>>(ws_cl, p_wenc, b_enc, ws_encl, 64, 25, 6);
  carafe2_k<<<cdiv(B_N * 25 * HWH, 256), 256, 0, stream>>>(ws_encl, ws_mli, ws_mlh, ws_mlr, B_N, 25, 64, 64);

  // 5) mask_lr_n (== ml_init2) -> output 0
  knorm_k<5><<<cdiv(B_N * HWH, 256), 256, 0, stream>>>(ws_mlr, out_mask, B_N, 128, 128);

  // 6) mask_hr = mask_hr_hr + carafe(conv3x3(cl,w_enc2), mask_lr_n, 5, 2) ; mask_hr_n
  conv_wmma_k<1, 3><<<gLR, 256, 0, stream>>>(ws_cl, p_wenc2, b_enc2, ws_enc2, 64, 9, 6);
  carafe2_k<<<cdiv(B_N * 9 * HWH, 256), 256, 0, stream>>>(ws_enc2, out_mask, ws_mhh, ws_mhr, B_N, 9, 64, 64);
  knorm_k<3><<<cdiv(B_N * HWH, 256), 256, 0, stream>>>(ws_mhr, ws_mhn, B_N, 128, 128);

  // 7) lr_up = carafe(lr_feat, mask_lr_n, 5, 2); hr_out = 2*hr - carafe(hr, mask_hr_n, 3, 1)
  carafe2_k<<<cdiv(B_N * 256 * HWH, 256), 256, 0, stream>>>(lr_feat, out_mask, nullptr, ws_lrup, B_N, 256, 64, 64);
  carafe1_res_k<<<cdiv(B_N * 256 * HWH, 256), 256, 0, stream>>>(hr_feat, ws_mhn, hr_feat, out_hr, B_N, 256, 128, 128);

  // 8) DySampler: group norms, similarities, offset convs (WMMA), sample.
  groupnorm_k<<<dim3(8, B_N), 256, 0, stream>>>(ws_ch2, gn_hr_w, gn_hr_b, ws_hrx, 64, HWH, 8);
  groupnorm_k<<<dim3(8, B_N), 256, 0, stream>>>(ws_cl, gn_lr_w, gn_lr_b, ws_lrx, 64, HWL, 8);
  similarity_k<<<cdiv(B_N * HWH, 256), 256, 0, stream>>>(ws_hrx, ws_hsim, B_N, 64, 128, 128);
  similarity_k<<<cdiv(B_N * HWL, 256), 256, 0, stream>>>(ws_lrx, ws_lsim, B_N, 64, 64, 64);

  conv_wmma_k<2, 3><<<gLR, 256, 0, stream>>>(ws_lsim, p_woff, b_off, ws_offl, 8, 32, 6);
  conv_wmma_k<1, 3><<<gHR, 256, 0, stream>>>(ws_hsim, p_whoff, b_hoff, ws_offh, 8, 8, 7);
  conv_wmma_k<2, 3><<<gLR, 256, 0, stream>>>(ws_lrx, p_wds, b_ds, ws_dsl, 64, 32, 6);
  conv_wmma_k<1, 3><<<gHR, 256, 0, stream>>>(ws_hrx, p_whds, b_hds, ws_dsh, 64, 8, 7);

  offset_k<<<cdiv(B_N * 32 * HWL, 256), 256, 0, stream>>>(ws_offl, ws_offh, ws_dsl, ws_dsh, ws_offs, B_N, 64, 64);
  dysample_k<<<cdiv(B_N * 4 * HWH, 256), 256, 0, stream>>>(ws_offs, ws_lrup, out_lr, B_N);
}
